// TimeSiries_82575041233490
// MI455X (gfx1250) — matrix operand order repaired and checked by reading, not compiled, not run on hardware
//
#include <hip/hip_runtime.h>
#include <hip/hip_bf16.h>

// ---------------- CDNA5 (gfx1250) types ----------------
typedef __attribute__((ext_vector_type(16))) _Float16 v16h;
typedef __attribute__((ext_vector_type(8)))  float    v8f;
typedef unsigned u32x4 __attribute__((ext_vector_type(4)));

#define HID    512
#define TENC   128
#define FDEC   32
#define TTOT   160
#define NBATCH 512
#define NWG    64      // 16 column-groups x 4 row-groups (persistent grid)
#define CGS    16

union ABPack { v16h v; u32x4 q[2]; };

__device__ __forceinline__ v8f zerov8() {
  v8f z = {0.f, 0.f, 0.f, 0.f, 0.f, 0.f, 0.f, 0.f};
  return z;
}

__device__ __forceinline__ float sigmoidf_(float x) {
  return 1.0f / (1.0f + __expf(-x));
}

// Device-wide sense-reversing barrier (requires all NWG workgroups resident).
__device__ __forceinline__ void grid_barrier(unsigned* cnt, unsigned* gen) {
  __threadfence();          // make this WG's global stores visible device-wide
  __syncthreads();
  if (threadIdx.x == 0) {
    unsigned g = __hip_atomic_load(gen, __ATOMIC_RELAXED, __HIP_MEMORY_SCOPE_AGENT);
    unsigned prev = __hip_atomic_fetch_add(cnt, 1u, __ATOMIC_ACQ_REL, __HIP_MEMORY_SCOPE_AGENT);
    if (prev == (unsigned)(NWG - 1)) {
      __hip_atomic_store(cnt, 0u, __ATOMIC_RELAXED, __HIP_MEMORY_SCOPE_AGENT);
      __hip_atomic_fetch_add(gen, 1u, __ATOMIC_ACQ_REL, __HIP_MEMORY_SCOPE_AGENT);
    } else {
      while (__hip_atomic_load(gen, __ATOMIC_ACQUIRE, __HIP_MEMORY_SCOPE_AGENT) == g) {
        __builtin_amdgcn_s_sleep(1);
      }
    }
  }
  __syncthreads();
}

// Accumulate acc[mt][g] += A[rows, k] * W[j(g), k]^T over K = HID, using
// v_wmma_f32_16x16x32_f16.  Per-lane addressing uses one shared k order for the 16-bit
// A(16x32)/B(32x16) operands, not AMD's table order: A and B lanes each hold 16 contiguous
// halves at k+{0|16}.
__device__ __forceinline__ void gemm_acc(
    const _Float16* __restrict__ A,     // [NBATCH][HID] row-major f16
    const _Float16* __restrict__ W,     // [4*HID][HID] row-major f16
    int arow0, int arow1,               // per-lane global rows of the 2 M tiles
    int j0, int j1, int j2, int j3,     // per-lane W rows of the 4 gate tiles
    int aoff, int boff,                 // per-lane K offsets (laneHalf based)
    v8f acc[2][4]) {
  const _Float16* a0p = A + (size_t)arow0 * HID + aoff;
  const _Float16* a1p = A + (size_t)arow1 * HID + aoff;
  const _Float16* b0p = W + (size_t)j0 * HID + boff;
  const _Float16* b1p = W + (size_t)j1 * HID + boff;
  const _Float16* b2p = W + (size_t)j2 * HID + boff;
  const _Float16* b3p = W + (size_t)j3 * HID + boff;
#pragma unroll 4
  for (int k0 = 0; k0 < HID; k0 += 32) {
    ABPack a0, a1, b0, b1, b2, b3;
    a0.q[0] = *(const u32x4*)(a0p + k0);
    a0.q[1] = *(const u32x4*)(a0p + k0 + 8);
    a1.q[0] = *(const u32x4*)(a1p + k0);
    a1.q[1] = *(const u32x4*)(a1p + k0 + 8);
    b0.q[0] = *(const u32x4*)(b0p + k0);
    b0.q[1] = *(const u32x4*)(b0p + k0 + 8);
    b1.q[0] = *(const u32x4*)(b1p + k0);
    b1.q[1] = *(const u32x4*)(b1p + k0 + 8);
    b2.q[0] = *(const u32x4*)(b2p + k0);
    b2.q[1] = *(const u32x4*)(b2p + k0 + 8);
    b3.q[0] = *(const u32x4*)(b3p + k0);
    b3.q[1] = *(const u32x4*)(b3p + k0 + 8);

    acc[0][0] = __builtin_amdgcn_wmma_f32_16x16x32_f16(false, a0.v, false, b0.v, (short)0, acc[0][0], false, false);
    acc[0][1] = __builtin_amdgcn_wmma_f32_16x16x32_f16(false, a0.v, false, b1.v, (short)0, acc[0][1], false, false);
    acc[0][2] = __builtin_amdgcn_wmma_f32_16x16x32_f16(false, a0.v, false, b2.v, (short)0, acc[0][2], false, false);
    acc[0][3] = __builtin_amdgcn_wmma_f32_16x16x32_f16(false, a0.v, false, b3.v, (short)0, acc[0][3], false, false);
    acc[1][0] = __builtin_amdgcn_wmma_f32_16x16x32_f16(false, a1.v, false, b0.v, (short)0, acc[1][0], false, false);
    acc[1][1] = __builtin_amdgcn_wmma_f32_16x16x32_f16(false, a1.v, false, b1.v, (short)0, acc[1][1], false, false);
    acc[1][2] = __builtin_amdgcn_wmma_f32_16x16x32_f16(false, a1.v, false, b2.v, (short)0, acc[1][2], false, false);
    acc[1][3] = __builtin_amdgcn_wmma_f32_16x16x32_f16(false, a1.v, false, b3.v, (short)0, acc[1][3], false, false);
  }
}

// --------- prep: f32->f16 weights, bias sums, zero state & barrier ctrl ----
__global__ void lstm_prep_kernel(
    const float* __restrict__ Whh1, const float* __restrict__ Wih2,
    const float* __restrict__ Whh2,
    const float* __restrict__ bih1, const float* __restrict__ bhh1,
    const float* __restrict__ bih2, const float* __restrict__ bhh2,
    _Float16* __restrict__ Whh1h, _Float16* __restrict__ Wih2h,
    _Float16* __restrict__ Whh2h,
    float* __restrict__ bsum1, float* __restrict__ bsum2,
    _Float16* __restrict__ h1, _Float16* __restrict__ h2,
    float* __restrict__ c1, float* __restrict__ c2,
    unsigned* __restrict__ ctrl) {
  const int tid = blockIdx.x * blockDim.x + threadIdx.x;
  const int stride = gridDim.x * blockDim.x;
  for (int k = tid; k < 4 * HID * HID; k += stride) {
    Whh1h[k] = (_Float16)Whh1[k];
    Wih2h[k] = (_Float16)Wih2[k];
    Whh2h[k] = (_Float16)Whh2[k];
  }
  for (int k = tid; k < 4 * HID; k += stride) {
    bsum1[k] = bih1[k] + bhh1[k];
    bsum2[k] = bih2[k] + bhh2[k];
  }
  for (int k = tid; k < NBATCH * HID; k += stride) {
    h1[k] = (_Float16)0.f;   // ping buffer 0 only (initial state)
    h2[k] = (_Float16)0.f;
    c1[k] = 0.f;
    c2[k] = 0.f;
  }
  if (tid < 2) ctrl[tid] = 0u;
}

// --------- persistent fused 2-layer LSTM + fc, WMMA f16 -------------------
__global__ __launch_bounds__(256, 1) void lstm_persistent_kernel(
    const float* __restrict__ x,        // [512][128]
    const float* __restrict__ targets,  // [512][32]
    const int*   __restrict__ tfmask,   // [32]
    const float* __restrict__ Wih1,     // [2048][1]
    const float* __restrict__ Wfc,      // [1][512]
    const float* __restrict__ bfc,      // [1]
    const _Float16* __restrict__ Whh1h,
    const _Float16* __restrict__ Wih2h,
    const _Float16* __restrict__ Whh2h,
    const float* __restrict__ bsum1, const float* __restrict__ bsum2,
    _Float16* __restrict__ h1, _Float16* __restrict__ h2,   // [2][512][512]
    float* __restrict__ c1, float* __restrict__ c2,         // [512][512]
    float* __restrict__ partial,        // [32][512]
    unsigned* __restrict__ ctrl,
    float* __restrict__ out)            // [512][160]
{
  __shared__ float outprev[128];        // previous-step fc output, this WG's rows

  const int cg = blockIdx.x & (CGS - 1);      // hidden-column group (32 units)
  const int rg = blockIdx.x >> 4;             // batch-row group (128 rows)
  const int rowBase = rg * 128;
  const int lane = threadIdx.x & 31;
  const int wave = threadIdx.x >> 5;
  const int r_w = wave & 3;                   // wave row position   (4)
  const int c_w = wave >> 2;                  // wave m-half position(2)
  const int lane15 = lane & 15;
  const int laneHalf = lane >> 4;

  const int aoff = laneHalf * 16;             // A k-offset (contiguous k order)
  const int boff = laneHalf * 16;             // B k-offset (same contiguous k order)
  const int arow0 = rowBase + (r_w * 2 + 0) * 16 + lane15;
  const int arow1 = rowBase + (r_w * 2 + 1) * 16 + lane15;
  const int mglob = cg * 32 + c_w * 16 + lane15;   // hidden unit this lane owns
  const int j0 = 0 * HID + mglob;             // i-gate weight row
  const int j1 = 1 * HID + mglob;             // f
  const int j2 = 2 * HID + mglob;             // g
  const int j3 = 3 * HID + mglob;             // o

  unsigned* cnt = ctrl;
  unsigned* gen = ctrl + 1;

  // Loop-invariant per-lane scalars.
  const float bfcv = bfc[0];
  const float wfcv = Wfc[mglob];
  const float bs1_0 = bsum1[j0], bs1_1 = bsum1[j1], bs1_2 = bsum1[j2], bs1_3 = bsum1[j3];
  const float bs2_0 = bsum2[j0], bs2_1 = bsum2[j1], bs2_2 = bsum2[j2], bs2_3 = bsum2[j3];
  const float wi1_0 = Wih1[j0], wi1_1 = Wih1[j1], wi1_2 = Wih1[j2], wi1_3 = Wih1[j3];

  for (int t = 0; ; ++t) {
    // ---- finish fc output of step t-1 (deterministic fixed-order sum) ----
    if (t >= 1) {
      if (threadIdx.x < 128) {
        const int row = rowBase + threadIdx.x;
        float s = bfcv;
#pragma unroll
        for (int p = 0; p < 32; ++p) s += partial[p * NBATCH + row];
        if (cg == 0) out[(size_t)row * TTOT + (t - 1)] = s;
        outprev[threadIdx.x] = s;
      }
      __syncthreads();
    }
    if (t == TTOT) break;

    const int cur = t & 1;
    const int nxt = cur ^ 1;
    const _Float16* h1cur = h1 + (size_t)cur * (NBATCH * HID);
    _Float16*       h1nxt = h1 + (size_t)nxt * (NBATCH * HID);
    const _Float16* h2cur = h2 + (size_t)cur * (NBATCH * HID);
    _Float16*       h2nxt = h2 + (size_t)nxt * (NBATCH * HID);

    // ================= layer 1: gates1 = h1 @ Whh1^T =====================
    v8f acc[2][4];
#pragma unroll
    for (int mt = 0; mt < 2; ++mt)
#pragma unroll
      for (int g = 0; g < 4; ++g) acc[mt][g] = zerov8();

    gemm_acc(h1cur, Whh1h, arow0, arow1, j0, j1, j2, j3, aoff, boff, acc);

    // cell 1 (register-local: all 4 gates of (n, mglob) sit in same lane)
#pragma unroll
    for (int mt = 0; mt < 2; ++mt) {
#pragma unroll
      for (int r = 0; r < 8; ++r) {
        const int n_l = (r_w * 2 + mt) * 16 + r + laneHalf * 8;
        const int n_g = rowBase + n_l;
        float xv;
        if (t < TENC) {
          xv = x[(size_t)n_g * TENC + t];
        } else {
          const int td = t - TENC;
          xv = (tfmask[td] > 0) ? targets[(size_t)n_g * FDEC + td]
                                : outprev[n_l];
        }
        const float ig = sigmoidf_(acc[mt][0][r] + bs1_0 + xv * wi1_0);
        const float fg = sigmoidf_(acc[mt][1][r] + bs1_1 + xv * wi1_1);
        const float gg = tanhf    (acc[mt][2][r] + bs1_2 + xv * wi1_2);
        const float og = sigmoidf_(acc[mt][3][r] + bs1_3 + xv * wi1_3);
        const size_t ci = (size_t)n_g * HID + mglob;
        const float cn = fg * c1[ci] + ig * gg;
        c1[ci] = cn;
        h1nxt[ci] = (_Float16)(og * tanhf(cn));
      }
    }

    grid_barrier(cnt, gen);   // h1[nxt] visible to all WGs

    // ========= layer 2: gates2 = h1_new @ Wih2^T + h2 @ Whh2^T ===========
#pragma unroll
    for (int mt = 0; mt < 2; ++mt)
#pragma unroll
      for (int g = 0; g < 4; ++g) acc[mt][g] = zerov8();

    gemm_acc(h1nxt, Wih2h, arow0, arow1, j0, j1, j2, j3, aoff, boff, acc);
    gemm_acc(h2cur, Whh2h, arow0, arow1, j0, j1, j2, j3, aoff, boff, acc);

    // cell 2 + fc partial (shfl butterfly over the 16 m-lanes)
#pragma unroll
    for (int mt = 0; mt < 2; ++mt) {
#pragma unroll
      for (int r = 0; r < 8; ++r) {
        const int n_l = (r_w * 2 + mt) * 16 + r + laneHalf * 8;
        const int n_g = rowBase + n_l;
        const float ig = sigmoidf_(acc[mt][0][r] + bs2_0);
        const float fg = sigmoidf_(acc[mt][1][r] + bs2_1);
        const float gg = tanhf    (acc[mt][2][r] + bs2_2);
        const float og = sigmoidf_(acc[mt][3][r] + bs2_3);
        const size_t ci = (size_t)n_g * HID + mglob;
        const float cn = fg * c2[ci] + ig * gg;
        c2[ci] = cn;
        const float hn = og * tanhf(cn);
        h2nxt[ci] = (_Float16)hn;

        float pv = hn * wfcv;                 // fc dot contribution
        pv += __shfl_xor(pv, 1, 32);
        pv += __shfl_xor(pv, 2, 32);
        pv += __shfl_xor(pv, 4, 32);
        pv += __shfl_xor(pv, 8, 32);          // sum over the 16 m-lanes
        if (lane15 == 0)
          partial[(cg * 2 + c_w) * NBATCH + n_g] = pv;
      }
    }

    grid_barrier(cnt, gen);   // h2[nxt] + partials visible to all WGs
  }
}

// --------------------------- host launcher --------------------------------
extern "C" void kernel_launch(void* const* d_in, const int* in_sizes, int n_in,
                              void* d_out, int out_size, void* d_ws, size_t ws_size,
                              hipStream_t stream) {
  (void)in_sizes; (void)n_in; (void)out_size; (void)ws_size;

  const float* x       = (const float*)d_in[0];
  const float* targets = (const float*)d_in[1];
  const int*   tfmask  = (const int*)d_in[2];
  // d_in[3] = future (compile-time constant 32)
  const float* Wih1 = (const float*)d_in[4];
  const float* Whh1 = (const float*)d_in[5];
  const float* bih1 = (const float*)d_in[6];
  const float* bhh1 = (const float*)d_in[7];
  const float* Wih2 = (const float*)d_in[8];
  const float* Whh2 = (const float*)d_in[9];
  const float* bih2 = (const float*)d_in[10];
  const float* bhh2 = (const float*)d_in[11];
  const float* Wfc  = (const float*)d_in[12];
  const float* bfc  = (const float*)d_in[13];
  float* out = (float*)d_out;

  char* ws = (char*)d_ws;
  constexpr size_t SZ_W  = (size_t)4 * HID * HID * sizeof(_Float16);   // 2 MB
  constexpr size_t SZ_H  = (size_t)2 * NBATCH * HID * sizeof(_Float16);// 1 MB
  constexpr size_t SZ_C  = (size_t)NBATCH * HID * sizeof(float);       // 1 MB
  constexpr size_t SZ_P  = (size_t)32 * NBATCH * sizeof(float);        // 64 KB
  constexpr size_t SZ_B  = (size_t)4 * HID * sizeof(float);            // 8 KB

  size_t o = 0;
  _Float16* Whh1h = (_Float16*)(ws + o); o += SZ_W;
  _Float16* Wih2h = (_Float16*)(ws + o); o += SZ_W;
  _Float16* Whh2h = (_Float16*)(ws + o); o += SZ_W;
  _Float16* h1    = (_Float16*)(ws + o); o += SZ_H;
  _Float16* h2    = (_Float16*)(ws + o); o += SZ_H;
  float*    c1    = (float*)(ws + o);    o += SZ_C;
  float*    c2    = (float*)(ws + o);    o += SZ_C;
  float*    part  = (float*)(ws + o);    o += SZ_P;
  float*    bsum1 = (float*)(ws + o);    o += SZ_B;
  float*    bsum2 = (float*)(ws + o);    o += SZ_B;
  unsigned* ctrl  = (unsigned*)(ws + o); o += 256;

  lstm_prep_kernel<<<dim3(512), dim3(256), 0, stream>>>(
      Whh1, Wih2, Whh2, bih1, bhh1, bih2, bhh2,
      Whh1h, Wih2h, Whh2h, bsum1, bsum2, h1, h2, c1, c2, ctrl);

  lstm_persistent_kernel<<<dim3(NWG), dim3(256), 0, stream>>>(
      x, targets, tfmask, Wih1, Wfc, bfc,
      Whh1h, Wih2h, Whh2h, bsum1, bsum2,
      h1, h2, c1, c2, part, ctrl, out);
}
